// InteractionGNN_48155173323295
// MI455X (gfx1250) — compile-verified, matrix operand order checked by reading, not run on hardware
//
#include <hip/hip_runtime.h>
#include <hip/hip_bf16.h>

// ---------------------------------------------------------------------------
// InteractionGNN for MI455X (gfx1250).
// - All 2-layer MLPs fused as wave32 WMMA (v_wmma_f32_16x16x32_f16) GEMMs
//   out of LDS-staged f16 tiles. Block = 128 threads (4 waves); tile = 64x128.
// - h/e activations kept in BOTH f32 (residual/scatter accuracy) and f16
//   (gather operand) form; edge gathers read the f16 copy -> half the HBM
//   bytes on the dominant traffic term.
// - Edge-kernel X staging uses GLOBAL_LOAD_ASYNC_TO_LDS_B128 (ASYNCcnt) so
//   the gather is a pure memory->LDS DMA with no VGPR round-trip.
// - Scatter-add aggregation reads e16 as b128 vectors, accumulates f32 atomics.
// ---------------------------------------------------------------------------

#define DHID 128

typedef __attribute__((ext_vector_type(16))) _Float16 v16h;
typedef __attribute__((ext_vector_type(8)))  _Float16 v8h;
typedef __attribute__((ext_vector_type(4)))  _Float16 v4h;
typedef __attribute__((ext_vector_type(8)))  float    v8f;

__device__ __forceinline__ void wait_async0() {
#if __has_builtin(__builtin_amdgcn_s_wait_asynccnt)
  __builtin_amdgcn_s_wait_asynccnt(0);
#else
  asm volatile("s_wait_asynccnt 0x0" ::: "memory");
#endif
}

// Issue one per-lane 16B async copy: LDS[ldsOff] = MEM[gaddr .. gaddr+15]
__device__ __forceinline__ void async_copy16(unsigned ldsOff, const _Float16* src) {
  asm volatile("global_load_async_to_lds_b128 %0, %1, off"
               :: "v"(ldsOff), "v"((unsigned long long)(uintptr_t)src)
               : "memory");
}

// A fragment (16x32 f16) per ISA layout: lane L(0-15): M=L, K = kb*32 + {0..7, 16..23};
// lane L+16: M=L, K = kb*32 + {8..15, 24..31}. Source is row-major [row][k] in LDS.
__device__ __forceinline__ v16h load_a_frag(const _Float16* Xs, int ldk, int wrow,
                                            int kb, int laneLow, int hiSel) {
  const _Float16* p = Xs + (wrow + laneLow) * ldk + kb * 32 + hiSel * 8;
  v8h lo = *(const v8h*)(p);        // K = base .. base+7
  v8h hi = *(const v8h*)(p + 16);   // K = base+16 .. base+23
  v16h a;
#pragma unroll
  for (int i = 0; i < 8; ++i) { a[i] = lo[i]; a[8 + i] = hi[i]; }
  return a;
}

// MODE: 0 = node encoder (X = nodes[row][0:16] f32, zero-padded to K=32)
//       1 = edge cat2    (X = [h16[start], h16[end]])            -- async DMA
//       2 = node cat     (X = [h[row] f32, msg[row] f32])
//       3 = edge cat3    (X = [h16[start], h16[end], e16[row]])  -- async DMA
// PRED: second layer is 128 -> 1 dot product (w2f f32, b2[0] scalar).
// O16 : epilogue also stores an f16 copy of the output.
template<int KIN, int MODE, bool RES, bool PRED, bool O16>
__launch_bounds__(128)
__global__ void mlp2_kernel(const float* __restrict__ srcA,
                            const float* __restrict__ srcB,
                            const _Float16* __restrict__ srcH16,
                            const _Float16* __restrict__ srcE16,
                            const int*   __restrict__ idxS,
                            const int*   __restrict__ idxE,
                            const _Float16* __restrict__ W1s,
                            const float* __restrict__ b1,
                            const _Float16* __restrict__ W2s,
                            const float* __restrict__ b2,
                            const float* __restrict__ w2f,
                            const float* __restrict__ resid,
                            float* __restrict__ out,
                            _Float16* __restrict__ out16, int M)
{
  constexpr int ROWS = 64;
  constexpr int NKB1 = KIN / 32;
  __shared__ _Float16 Xs[ROWS * KIN];
  __shared__ _Float16 Hs[ROWS * DHID];
  __shared__ int sS[ROWS], sE[ROWS];

  const int tid = threadIdx.x;
  const int rowBase = blockIdx.x * ROWS;

  if (MODE == 1 || MODE == 3) {
    if (tid < ROWS) {
      int r = rowBase + tid;
      sS[tid] = (r < M) ? idxS[r] : 0;
    } else if (tid < 2 * ROWS) {
      int r = rowBase + (tid - ROWS);
      sE[tid - ROWS] = (r < M) ? idxE[r] : 0;
    }
    __syncthreads();
  }

  if (MODE == 1 || MODE == 3) {
    // ---- pure-f16 gather: async memory->LDS, 16B per lane per issue ----
    constexpr int C8 = KIN / 8;
    for (int g = tid; g < ROWS * C8; g += 128) {
      int r = g / C8;
      int c = (g % C8) * 8;
      int gr  = rowBase + r;
      int grc = (gr < M) ? gr : (M - 1);
      const _Float16* src;
      if (MODE == 1) {
        src = (c < DHID) ? srcH16 + (size_t)sS[r] * DHID + c
                         : srcH16 + (size_t)sE[r] * DHID + (c - DHID);
      } else {
        if (c < DHID)          src = srcH16 + (size_t)sS[r] * DHID + c;
        else if (c < 2 * DHID) src = srcH16 + (size_t)sE[r] * DHID + (c - DHID);
        else                   src = srcE16 + (size_t)grc * DHID + (c - 2 * DHID);
      }
      unsigned ldsOff = (unsigned)(uintptr_t)(Xs + r * KIN + c);
      async_copy16(ldsOff, src);
    }
    wait_async0();
  } else {
    // ---- f32 sources: load, convert to f16, store to LDS ----
    constexpr int G = ROWS * KIN / 4;
    for (int g = tid; g < G; g += 128) {
      int r = g / (KIN / 4);
      int c = (g % (KIN / 4)) * 4;
      int gr = rowBase + r;
      float4 v = make_float4(0.f, 0.f, 0.f, 0.f);
      if (gr < M) {
        if (MODE == 0) {
          if (c < 16) v = *(const float4*)(srcA + (size_t)gr * 16 + c);
        } else { // MODE 2
          v = (c < DHID) ? *(const float4*)(srcA + (size_t)gr * DHID + c)
                         : *(const float4*)(srcB + (size_t)gr * DHID + (c - DHID));
        }
      }
      v4h hv = { (_Float16)v.x, (_Float16)v.y, (_Float16)v.z, (_Float16)v.w };
      *(v4h*)(Xs + r * KIN + c) = hv;
    }
  }
  __syncthreads();

  const int lane    = tid & 31;
  const int wave    = tid >> 5;
  const int wrow    = wave * 16;
  const int laneLow = lane & 15;
  const int hiSel   = lane >> 4;

  const v8f zero8 = {0.f, 0.f, 0.f, 0.f, 0.f, 0.f, 0.f, 0.f};
  v8f acc[8];
#pragma unroll
  for (int t = 0; t < 8; ++t) acc[t] = zero8;

  // ---- GEMM1: hidden = relu(X @ W1 + b1) ----
  for (int kb = 0; kb < NKB1; ++kb) {
    v16h a = load_a_frag(Xs, KIN, wrow, kb, laneLow, hiSel);
#pragma unroll
    for (int t = 0; t < 8; ++t) {
      v16h b = *(const v16h*)(W1s + ((size_t)(t * NKB1 + kb) * 32 + lane) * 16);
      acc[t] = __builtin_amdgcn_wmma_f32_16x16x32_f16(
          false, a, false, b, (short)0, acc[t], false, false);
    }
  }

#pragma unroll
  for (int t = 0; t < 8; ++t) {
    float bv = b1[t * 16 + laneLow];
#pragma unroll
    for (int r = 0; r < 8; ++r) {
      float x = acc[t][r] + bv;
      x = x > 0.f ? x : 0.f;
      Hs[(wrow + r + 8 * hiSel) * DHID + t * 16 + laneLow] = (_Float16)x;
    }
  }
  __syncthreads();

  if (PRED) {
    // ---- layer 2 is 128 -> 1: per-row dot with w2f ----
    if (tid < ROWS) {
      int gr = rowBase + tid;
      if (gr < M) {
        float s = b2[0];
        const _Float16* hr = Hs + tid * DHID;
#pragma unroll 8
        for (int n = 0; n < DHID; ++n) s += (float)hr[n] * w2f[n];
        out[gr] = s;
      }
    }
    return;
  }

  // ---- GEMM2: out = hidden @ W2 + b2 (+ residual) ----
#pragma unroll
  for (int t = 0; t < 8; ++t) acc[t] = zero8;
  for (int kb = 0; kb < DHID / 32; ++kb) {
    v16h a = load_a_frag(Hs, DHID, wrow, kb, laneLow, hiSel);
#pragma unroll
    for (int t = 0; t < 8; ++t) {
      v16h b = *(const v16h*)(W2s + ((size_t)(t * (DHID / 32) + kb) * 32 + lane) * 16);
      acc[t] = __builtin_amdgcn_wmma_f32_16x16x32_f16(
          false, a, false, b, (short)0, acc[t], false, false);
    }
  }

#pragma unroll
  for (int t = 0; t < 8; ++t) {
    float bv = b2[t * 16 + laneLow];
#pragma unroll
    for (int r = 0; r < 8; ++r) {
      int gr = rowBase + wrow + r + 8 * hiSel;
      if (gr < M) {
        int n = t * 16 + laneLow;
        float x = acc[t][r] + bv;
        if (RES) x += resid[(size_t)gr * DHID + n];
        out[(size_t)gr * DHID + n] = x;
        if (O16) out16[(size_t)gr * DHID + n] = (_Float16)x;
      }
    }
  }
}

// Convert W[K][128] f32 -> f16 fragment-swizzled (Kp padded, zero fill).
// Per-lane B fragment = 16 contiguous halves: lane = nl + 16*((kr>>3)&1),
// halfIdx = ((kr>>4)<<3) + (kr&7)  (mirrors the A-fragment K striping).
__global__ void swizzle_w_kernel(const float* __restrict__ W,
                                 _Float16* __restrict__ dst, int K, int Kp)
{
  int i = blockIdx.x * blockDim.x + threadIdx.x;
  int total = Kp * 128;
  if (i >= total) return;
  int k = i >> 7;
  int n = i & 127;
  float v = (k < K) ? W[(size_t)k * 128 + n] : 0.f;
  int nKb   = Kp >> 5;
  int ntile = n >> 4, nl = n & 15;
  int kb = k >> 5, kr = k & 31;
  int lane    = nl + (((kr >> 3) & 1) << 4);
  int halfIdx = ((kr >> 4) << 3) + (kr & 7);
  dst[((size_t)(ntile * nKb + kb) * 32 + lane) * 16 + halfIdx] = (_Float16)v;
}

__global__ void zero_f32v4_kernel(float4* __restrict__ p, long long n4) {
  long long i = (long long)blockIdx.x * blockDim.x + threadIdx.x;
  if (i < n4) p[i] = make_float4(0.f, 0.f, 0.f, 0.f);
}

// msg[end[e]] += e_feat[e]: b128 f16 source reads, f32 atomic accumulation.
// One thread per 8 elements -> 16 threads per edge.
__global__ void aggregate_kernel(const _Float16* __restrict__ e16,
                                 const int* __restrict__ idxE,
                                 float* __restrict__ msg, int E)
{
  long long i = (long long)blockIdx.x * blockDim.x + threadIdx.x;
  if (i >= (long long)E * (DHID / 8)) return;
  int e = (int)(i >> 4);          // DHID/8 = 16 groups per edge
  int c = (int)(i & 15) * 8;
  v8h v = *(const v8h*)(e16 + (size_t)e * DHID + c);
  float* dst = msg + (size_t)idxE[e] * DHID + c;
#pragma unroll
  for (int j = 0; j < 8; ++j) atomicAdd(dst + j, (float)v[j]);
}

extern "C" void kernel_launch(void* const* d_in, const int* in_sizes, int n_in,
                              void* d_out, int out_size, void* d_ws, size_t ws_size,
                              hipStream_t stream) {
  (void)n_in; (void)ws_size;
  const float* nodes = (const float*)d_in[0];
  const int*   eidx  = (const int*)d_in[1];
  const float* ne_W1 = (const float*)d_in[2];  const float* ne_b1 = (const float*)d_in[3];
  const float* ne_W2 = (const float*)d_in[4];  const float* ne_b2 = (const float*)d_in[5];
  const float* ee_W1 = (const float*)d_in[6];  const float* ee_b1 = (const float*)d_in[7];
  const float* ee_W2 = (const float*)d_in[8];  const float* ee_b2 = (const float*)d_in[9];
  const float* nn_W1 = (const float*)d_in[10]; const float* nn_b1 = (const float*)d_in[11];
  const float* nn_W2 = (const float*)d_in[12]; const float* nn_b2 = (const float*)d_in[13];
  const float* en_W1 = (const float*)d_in[14]; const float* en_b1 = (const float*)d_in[15];
  const float* en_W2 = (const float*)d_in[16]; const float* en_b2 = (const float*)d_in[17];
  const float* pe_W1 = (const float*)d_in[18]; const float* pe_b1 = (const float*)d_in[19];
  const float* pe_W2 = (const float*)d_in[20]; const float* pe_b2 = (const float*)d_in[21];

  const int N = in_sizes[0] / 16;
  const int E = out_size;
  const int* idxS = eidx;
  const int* idxE = eidx + E;

  // ---- workspace layout ----
  char* ws = (char*)d_ws;
  size_t offh   = (size_t)N * DHID * sizeof(float);
  size_t offe   = (size_t)E * DHID * sizeof(float);
  size_t offh16 = (size_t)N * DHID * sizeof(_Float16);
  size_t offe16 = (size_t)E * DHID * sizeof(_Float16);
  size_t cur = 0;
  float* hA   = (float*)(ws + cur); cur += offh;
  float* hB   = (float*)(ws + cur); cur += offh;
  float* msg  = (float*)(ws + cur); cur += offh;
  float* eA   = (float*)(ws + cur); cur += offe;
  float* eB   = (float*)(ws + cur); cur += offe;
  _Float16* hA16 = (_Float16*)(ws + cur); cur += offh16;
  _Float16* hB16 = (_Float16*)(ws + cur); cur += offh16;
  _Float16* eA16 = (_Float16*)(ws + cur); cur += offe16;
  _Float16* eB16 = (_Float16*)(ws + cur); cur += offe16;
  _Float16* wf   = (_Float16*)(ws + cur);

  _Float16* ne1s = wf + 0;        // 32  x128
  _Float16* ne2s = wf + 4096;     // 128 x128
  _Float16* ee1s = wf + 20480;    // 256 x128
  _Float16* ee2s = wf + 53248;    // 128 x128
  _Float16* nn1s = wf + 69632;    // 256 x128
  _Float16* nn2s = wf + 102400;   // 128 x128
  _Float16* en1s = wf + 118784;   // 384 x128
  _Float16* en2s = wf + 167936;   // 128 x128
  _Float16* pe1s = wf + 184320;   // 384 x128

  auto sw = [&](const float* W, _Float16* dst, int K, int Kp) {
    int total = Kp * 128;
    swizzle_w_kernel<<<(total + 255) / 256, 256, 0, stream>>>(W, dst, K, Kp);
  };
  sw(ne_W1, ne1s, 16, 32);
  sw(ne_W2, ne2s, 128, 128);
  sw(ee_W1, ee1s, 256, 256);
  sw(ee_W2, ee2s, 128, 128);
  sw(nn_W1, nn1s, 256, 256);
  sw(nn_W2, nn2s, 128, 128);
  sw(en_W1, en1s, 384, 384);
  sw(en_W2, en2s, 128, 128);
  sw(pe_W1, pe1s, 384, 384);

  const int nTiles = (N + 63) / 64;
  const int eTiles = (E + 63) / 64;

  // h = MLP_ne(nodes)   (also emits h in f16)
  mlp2_kernel<32, 0, false, false, true><<<nTiles, 128, 0, stream>>>(
      nodes, nullptr, nullptr, nullptr, nullptr, nullptr,
      ne1s, ne_b1, ne2s, ne_b2, nullptr, nullptr, hA, hA16, N);

  // e = MLP_ee([h[s], h[d]])   (f16 gathers via async-to-LDS)
  mlp2_kernel<256, 1, false, false, true><<<eTiles, 128, 0, stream>>>(
      nullptr, nullptr, hA16, nullptr, idxS, idxE,
      ee1s, ee_b1, ee2s, ee_b2, nullptr, nullptr, eA, eA16, E);

  float* hc = hA; float* hn = hB;
  float* ec = eA; float* en = eB;
  _Float16* hc16 = hA16; _Float16* hn16 = hB16;
  _Float16* ec16 = eA16; _Float16* en16 = eB16;
  long long nGrp4N = (long long)N * DHID / 4;
  long long nGrp8E = (long long)E * (DHID / 8);

  for (int it = 0; it < 4; ++it) {
    zero_f32v4_kernel<<<(unsigned)((nGrp4N + 255) / 256), 256, 0, stream>>>(
        (float4*)msg, nGrp4N);
    aggregate_kernel<<<(unsigned)((nGrp8E + 255) / 256), 256, 0, stream>>>(
        ec16, idxE, msg, E);

    // h_new = MLP_nn([h, msg]) + h
    mlp2_kernel<256, 2, true, false, true><<<nTiles, 128, 0, stream>>>(
        hc, msg, nullptr, nullptr, nullptr, nullptr,
        nn1s, nn_b1, nn2s, nn_b2, nullptr, hc, hn, hn16, N);

    // e_new = MLP_en([h[s], h[d], e]) + e   (uses old h; f16 async gathers)
    mlp2_kernel<384, 3, true, false, true><<<eTiles, 128, 0, stream>>>(
        nullptr, nullptr, hc16, ec16, idxS, idxE,
        en1s, en_b1, en2s, en_b2, nullptr, ec, en, en16, E);

    float* t;
    t = hc; hc = hn; hn = t;
    t = ec; ec = en; en = t;
    _Float16* t16;
    t16 = hc16; hc16 = hn16; hn16 = t16;
    t16 = ec16; ec16 = en16; en16 = t16;
  }

  // pred = MLP_pe([h[s], h[d], e])[:, 0]
  mlp2_kernel<384, 3, false, true, false><<<eTiles, 128, 0, stream>>>(
      nullptr, nullptr, hc16, ec16, idxS, idxE,
      pe1s, pe_b1, nullptr, pe_b2, pe_W2, nullptr, (float*)d_out, nullptr, E);
}
